// LambdaNeuralNetwork_36902359007812
// MI455X (gfx1250) — matrix operand order repaired and checked by reading, not compiled, not run on hardware
//
#include <hip/hip_runtime.h>

typedef __attribute__((ext_vector_type(16))) _Float16 v16h;
typedef __attribute__((ext_vector_type(8)))  _Float16 v8h;
typedef __attribute__((ext_vector_type(8)))  float    v8f;

// ---------- helpers ----------
__device__ __forceinline__ unsigned fenc(float f) {
  unsigned u = __float_as_uint(f);
  return (u & 0x80000000u) ? ~u : (u | 0x80000000u);   // order-preserving map
}
__device__ __forceinline__ float fdec(unsigned e) {
  return __uint_as_float((e & 0x80000000u) ? (e & 0x7FFFFFFFu) : ~e);
}
__device__ __forceinline__ unsigned umn(unsigned a, unsigned b){ return a < b ? a : b; }
__device__ __forceinline__ unsigned umx(unsigned a, unsigned b){ return a > b ? a : b; }

// ---------- utility kernels ----------
__global__ void k_zero_f32(float* p, int n) {
  int t = blockIdx.x * blockDim.x + threadIdx.x;
  if (t < n) p[t] = 0.f;
}

__global__ void k_init_small(float* acc, unsigned* mm) {
  int t = threadIdx.x;
  if (t < 65) acc[t] = 0.f;                       // acc[0..63] sums, acc[64] count
  if (t == 65) { mm[0] = 0xFFFFFFFFu; mm[1] = 0u; } // min-enc, max-enc identities
}

__global__ void k_cvt_f16(const float* __restrict__ s, _Float16* __restrict__ d, int n) {
  int t = blockIdx.x * blockDim.x + threadIdx.x;
  if (t < n) d[t] = (_Float16)s[t];
}

// ---------- timestamp min/max ----------
__global__ void k_minmax(const float* __restrict__ ts, unsigned* mm, int N) {
  __shared__ unsigned smn[256], smx[256];
  int t = threadIdx.x;
  unsigned mn = 0xFFFFFFFFu, mx = 0u;
  for (int i = blockIdx.x * blockDim.x + t; i < N; i += gridDim.x * blockDim.x) {
    unsigned e = fenc(ts[i]);
    mn = umn(mn, e); mx = umx(mx, e);
  }
  smn[t] = mn; smx[t] = mx;
  __syncthreads();
  for (int s = 128; s > 0; s >>= 1) {
    if (t < s) { smn[t] = umn(smn[t], smn[t + s]); smx[t] = umx(smx[t], smx[t + s]); }
    __syncthreads();
  }
  if (t == 0) { atomicMin(&mm[0], smn[0]); atomicMax(&mm[1], smx[0]); }
}

// ---------- h0 = concat(x, relu(t_norm * W_temp^T + b_temp)) as f16 ----------
__global__ void k_build_h0(const float* __restrict__ x, const float* __restrict__ ts,
                           const unsigned* __restrict__ mm,
                           const float* __restrict__ Wt, const float* __restrict__ bt,
                           _Float16* __restrict__ h0, int N, int FIN, int T4) {
  int n = blockIdx.x * blockDim.x + threadIdx.x;
  if (n >= N) return;
  float mn = fdec(mm[0]), mx = fdec(mm[1]);
  float t = (ts[n] - mn) / (mx - mn + 1e-8f);
  _Float16* row = h0 + (size_t)n * (FIN + T4);
  for (int j = 0; j < FIN; ++j) row[j] = (_Float16)x[(size_t)n * FIN + j];
  for (int j = 0; j < T4; ++j) row[FIN + j] = (_Float16)fmaxf(t * Wt[j] + bt[j], 0.f);
}

// ---------- WMMA GEMM: G[N,256] = H[N,K](f16) x W[256,K](f16)^T, f32 accum ----------
__global__ __launch_bounds__(256) void k_gemm_wmma(
    const _Float16* __restrict__ H, const _Float16* __restrict__ W,
    float* __restrict__ G, int N, int K, int mtiles) {
  int gid  = blockIdx.x * blockDim.x + threadIdx.x;
  int wave = gid >> 5;                // wave32
  int lane = threadIdx.x & 31;
  if (wave >= mtiles * 16) return;    // wave-uniform guard (EXEC stays all-1s)
  int jt = wave & 15;                 // 256/16 = 16 column tiles
  int mt = wave >> 4;
  int ln = lane & 15;
  int hh = lane >> 4;

  int ar = mt * 16 + ln; if (ar >= N) ar = N - 1;      // clamp (stores are guarded)
  const _Float16* arow = H + (size_t)ar * K;
  const _Float16* brow = W + (size_t)(jt * 16 + ln) * K;

  v8f c = {};
  for (int k0 = 0; k0 < K; k0 += 32) {
    union { v16h v; v8h h2[2]; } a, b;
    // A 16x32 f16 layout: lane half hh -> K in [8hh,8hh+8) then [16+8hh,16+8hh+8)
    a.h2[0] = *(const v8h*)(arow + k0 + 8 * hh);
    a.h2[1] = *(const v8h*)(arow + k0 + 16 + 8 * hh);
    // B 32x16 f16 layout: lane half hh -> K in [8hh,8hh+8) then [16+8hh,16+8hh+8), col = ln
    b.h2[0] = *(const v8h*)(brow + k0 + 8 * hh);
    b.h2[1] = *(const v8h*)(brow + k0 + 16 + 8 * hh);
    c = __builtin_amdgcn_wmma_f32_16x16x32_f16(false, a.v, false, b.v,
                                               (short)0, c, false, false);
  }
  int col = jt * 16 + ln;
  #pragma unroll
  for (int r = 0; r < 8; ++r) {
    int row = mt * 16 + 8 * hh + r;   // C/D layout: VGPR r -> M = r + 8*half
    if (row < N) G[(size_t)row * 256 + col] = c[r];
  }
}

// ---------- attention logits: al[n,h] = dot(g[n,h,:], a[h,:]) ----------
__global__ void k_alogits(const float* __restrict__ g,
                          const float* __restrict__ asrc, const float* __restrict__ adst,
                          float* __restrict__ alsrc, float* __restrict__ aldst, int N) {
  int t = blockIdx.x * blockDim.x + threadIdx.x;
  if (t >= N * 4) return;
  int n = t >> 2, h = t & 3;
  const float* grow = g + (size_t)n * 256 + h * 64;
  const float* as = asrc + h * 64;
  const float* ad = adst + h * 64;
  float s0 = 0.f, s1 = 0.f;
  #pragma unroll 8
  for (int c = 0; c < 64; ++c) { float v = grow[c]; s0 += v * as[c]; s1 += v * ad[c]; }
  alsrc[t] = s0; aldst[t] = s1;
}

// ---------- edge pass 1: e = leaky_relu(..); segment max ----------
__global__ void k_edge1(const int* __restrict__ he, int E, int EN,
                        const float* __restrict__ alsrc, const float* __restrict__ aldst,
                        float* __restrict__ ebuf, unsigned* __restrict__ menc) {
  int t = blockIdx.x * blockDim.x + threadIdx.x;
  if (t >= EN * 4) return;
  int e = t >> 2, h = t & 3;
  int src, dst;
  if (e < E) { src = he[e]; dst = he[E + e]; } else { src = dst = e - E; }
  float v = alsrc[src * 4 + h] + aldst[dst * 4 + h];
  v = v > 0.f ? v : 0.2f * v;
  ebuf[t] = v;
  atomicMax(&menc[dst * 4 + h], fenc(v));
}

// ---------- edge pass 2: ex = exp(e - m[dst]); segment sum ----------
__global__ void k_edge2(const int* __restrict__ he, int E, int EN,
                        float* __restrict__ ebuf, const unsigned* __restrict__ menc,
                        float* __restrict__ ssum) {
  int t = blockIdx.x * blockDim.x + threadIdx.x;
  if (t >= EN * 4) return;
  int e = t >> 2, h = t & 3;
  int dst = (e < E) ? he[E + e] : (e - E);
  float ex = __expf(ebuf[t] - fdec(menc[dst * 4 + h]));
  ebuf[t] = ex;
  atomicAdd(&ssum[dst * 4 + h], ex);
}

// ---------- edge pass 3: out[dst,:] += g[src,:] * alpha ----------
__global__ void k_edge3(const int* __restrict__ he, int E, long long EN,
                        const float* __restrict__ ebuf, const float* __restrict__ ssum,
                        const float* __restrict__ g, float* __restrict__ outacc) {
  long long t = (long long)blockIdx.x * blockDim.x + threadIdx.x;
  if (t >= EN * 256) return;
  int e = (int)(t >> 8);
  int c = (int)(t & 255);
  int h = c >> 6;
  int src, dst;
  if (e < E) { src = he[e]; dst = he[E + e]; } else { src = dst = e - E; }
  float alpha = ebuf[e * 4 + h] / (ssum[dst * 4 + h] + 1e-16f);
  atomicAdd(&outacc[(size_t)dst * 256 + c], g[(size_t)src * 256 + c] * alpha);
}

// ---------- head mean + bias (+relu), f32 and optional f16 outputs ----------
__global__ void k_finalize(const float* __restrict__ outacc, const float* __restrict__ bias,
                           float* __restrict__ of32, _Float16* of16, int N, int dorelu) {
  int t = blockIdx.x * blockDim.x + threadIdx.x;
  if (t >= N * 64) return;
  int n = t >> 6, c = t & 63;
  const float* row = outacc + (size_t)n * 256;
  float v = 0.25f * (row[c] + row[64 + c] + row[128 + c] + row[192 + c]) + bias[c];
  if (dorelu) v = fmaxf(v, 0.f);
  of32[t] = v;
  if (of16) of16[t] = (_Float16)v;
}

// ---------- realtime mask / masked mean ----------
__global__ void k_set_mask(const int* __restrict__ rte, float* __restrict__ mask, int RT) {
  int t = blockIdx.x * blockDim.x + threadIdx.x;
  if (t < RT) mask[rte[t]] = 1.f;
}

__global__ void k_rt_reduce(const float* __restrict__ bemb, const float* __restrict__ mask,
                            float* __restrict__ acc, int N) {
  int n = blockIdx.x * blockDim.x + threadIdx.x;
  if (n >= N) return;
  if (mask[n] != 0.f) {
    atomicAdd(&acc[64], 1.f);
    for (int c = 0; c < 64; ++c) atomicAdd(&acc[c], bemb[(size_t)n * 64 + c]);
  }
}

// ---------- fraud head MLP (single block) ----------
__global__ void k_head(const float* __restrict__ acc,
                       const float* __restrict__ W_rt, const float* __restrict__ b_rt,
                       const float* __restrict__ Wf1, const float* __restrict__ bf1,
                       const float* __restrict__ Wf2, const float* __restrict__ bf2,
                       const float* __restrict__ Wf3, const float* __restrict__ bf3,
                       float* __restrict__ out) {
  __shared__ float comb[128];
  __shared__ float z1[64];
  __shared__ float z2[32];
  int t = threadIdx.x;
  float cnt = acc[64];
  if (t < 64) comb[t] = acc[t] / cnt;              // rt_context
  __syncthreads();
  if (t < 64) {                                    // rt_emb = relu(W_rt @ ctx + b_rt)
    float s = b_rt[t];
    for (int k = 0; k < 64; ++k) s += W_rt[t * 64 + k] * comb[k];
    comb[64 + t] = fmaxf(s, 0.f);
  }
  __syncthreads();
  if (t < 64) {                                    // z1 = relu(Wf1 @ combined + bf1)
    float s = bf1[t];
    for (int k = 0; k < 128; ++k) s += Wf1[t * 128 + k] * comb[k];
    z1[t] = fmaxf(s, 0.f);
  }
  __syncthreads();
  if (t < 32) {                                    // z2 = relu(Wf2 @ z1 + bf2)
    float s = bf2[t];
    for (int k = 0; k < 64; ++k) s += Wf2[t * 64 + k] * z1[k];
    z2[t] = fmaxf(s, 0.f);
  }
  __syncthreads();
  if (t == 0) {                                    // sigmoid(Wf3 @ z2 + bf3)
    float s = bf3[0];
    for (int k = 0; k < 32; ++k) s += Wf3[k] * z2[k];
    out[0] = 1.f / (1.f + expf(-s));
  }
}

// =====================================================================
extern "C" void kernel_launch(void* const* d_in, const int* in_sizes, int n_in,
                              void* d_out, int out_size, void* d_ws, size_t ws_size,
                              hipStream_t stream) {
  const float* x      = (const float*)d_in[0];
  const int*   he     = (const int*)d_in[1];
  const int*   rte    = (const int*)d_in[2];
  const float* ts     = (const float*)d_in[3];
  const float* W_temp = (const float*)d_in[4];
  const float* b_temp = (const float*)d_in[5];
  const float* W1     = (const float*)d_in[6];
  const float* a1_src = (const float*)d_in[7];
  const float* a1_dst = (const float*)d_in[8];
  const float* bias1  = (const float*)d_in[9];
  const float* W2     = (const float*)d_in[10];
  const float* a2_src = (const float*)d_in[11];
  const float* a2_dst = (const float*)d_in[12];
  const float* bias2  = (const float*)d_in[13];
  const float* W_rt   = (const float*)d_in[14];
  const float* b_rt   = (const float*)d_in[15];
  const float* Wf1    = (const float*)d_in[16];
  const float* bf1    = (const float*)d_in[17];
  const float* Wf2    = (const float*)d_in[18];
  const float* bf2    = (const float*)d_in[19];
  const float* Wf3    = (const float*)d_in[20];
  const float* bf3    = (const float*)d_in[21];
  float* out = (float*)d_out;

  const int N   = in_sizes[3];
  const int E   = in_sizes[1] / 2;
  const int EN  = E + N;
  const int RT  = in_sizes[2];
  const int FIN = in_sizes[0] / N;        // 16
  const int T4  = in_sizes[4];            // 16
  const int K1  = in_sizes[6]  / 256;     // 32
  const int K2  = in_sizes[10] / 256;     // 64
  const int mtiles = (N + 15) / 16;

  // ---- workspace carve-out ----
  char* wp = (char*)d_ws;
  size_t off = 0;
  auto take = [&](size_t bytes) -> char* {
    char* p = wp + off;
    off += (bytes + 255) & ~(size_t)255;
    return p;
  };
  float*    g_buf  = (float*)take((size_t)N * 256 * 4);
  float*    outacc = (float*)take((size_t)N * 256 * 4);
  float*    ebuf   = (float*)take((size_t)EN * 4 * 4);
  float*    alsrc  = (float*)take((size_t)N * 4 * 4);
  float*    aldst  = (float*)take((size_t)N * 4 * 4);
  unsigned* menc   = (unsigned*)take((size_t)N * 4 * 4);
  float*    ssum   = (float*)take((size_t)N * 4 * 4);
  float*    h1f32  = (float*)take((size_t)N * 64 * 4);
  float*    bemb   = (float*)take((size_t)N * 64 * 4);
  _Float16* h0f16  = (_Float16*)take((size_t)N * (FIN + T4) * 2);
  _Float16* h1f16  = (_Float16*)take((size_t)N * 64 * 2);
  _Float16* W1h    = (_Float16*)take((size_t)in_sizes[6] * 2);
  _Float16* W2h    = (_Float16*)take((size_t)in_sizes[10] * 2);
  float*    mask   = (float*)take((size_t)N * 4);
  float*    acc    = (float*)take(65 * 4);
  unsigned* mm     = (unsigned*)take(2 * 4);

  auto gr = [](long long n) { return (unsigned)((n + 255) / 256); };

  // ---- setup ----
  k_init_small<<<1, 128, 0, stream>>>(acc, mm);
  k_zero_f32<<<gr(N), 256, 0, stream>>>(mask, N);
  k_minmax<<<256, 256, 0, stream>>>(ts, mm, N);
  k_cvt_f16<<<gr(in_sizes[6]), 256, 0, stream>>>(W1, W1h, in_sizes[6]);
  k_cvt_f16<<<gr(in_sizes[10]), 256, 0, stream>>>(W2, W2h, in_sizes[10]);
  k_build_h0<<<gr(N), 256, 0, stream>>>(x, ts, mm, W_temp, b_temp, h0f16, N, FIN, T4);

  const unsigned gemm_blocks = gr((long long)mtiles * 16 * 32);

  // ---- GAT layer 1 ----
  k_zero_f32<<<gr((long long)N * 256), 256, 0, stream>>>(outacc, N * 256);
  k_zero_f32<<<gr((long long)N * 4), 256, 0, stream>>>((float*)menc, N * 4);
  k_zero_f32<<<gr((long long)N * 4), 256, 0, stream>>>(ssum, N * 4);
  k_gemm_wmma<<<gemm_blocks, 256, 0, stream>>>(h0f16, W1h, g_buf, N, K1, mtiles);
  k_alogits<<<gr((long long)N * 4), 256, 0, stream>>>(g_buf, a1_src, a1_dst, alsrc, aldst, N);
  k_edge1<<<gr((long long)EN * 4), 256, 0, stream>>>(he, E, EN, alsrc, aldst, ebuf, menc);
  k_edge2<<<gr((long long)EN * 4), 256, 0, stream>>>(he, E, EN, ebuf, menc, ssum);
  k_edge3<<<gr((long long)EN * 256), 256, 0, stream>>>(he, E, EN, ebuf, ssum, g_buf, outacc);
  k_finalize<<<gr((long long)N * 64), 256, 0, stream>>>(outacc, bias1, h1f32, h1f16, N, 1);

  // ---- GAT layer 2 ----
  k_zero_f32<<<gr((long long)N * 256), 256, 0, stream>>>(outacc, N * 256);
  k_zero_f32<<<gr((long long)N * 4), 256, 0, stream>>>((float*)menc, N * 4);
  k_zero_f32<<<gr((long long)N * 4), 256, 0, stream>>>(ssum, N * 4);
  k_gemm_wmma<<<gemm_blocks, 256, 0, stream>>>(h1f16, W2h, g_buf, N, K2, mtiles);
  k_alogits<<<gr((long long)N * 4), 256, 0, stream>>>(g_buf, a2_src, a2_dst, alsrc, aldst, N);
  k_edge1<<<gr((long long)EN * 4), 256, 0, stream>>>(he, E, EN, alsrc, aldst, ebuf, menc);
  k_edge2<<<gr((long long)EN * 4), 256, 0, stream>>>(he, E, EN, ebuf, menc, ssum);
  k_edge3<<<gr((long long)EN * 256), 256, 0, stream>>>(he, E, EN, ebuf, ssum, g_buf, outacc);
  k_finalize<<<gr((long long)N * 64), 256, 0, stream>>>(outacc, bias2, bemb, (_Float16*)nullptr, N, 0);

  // ---- realtime context + fraud head ----
  k_set_mask<<<gr(RT), 256, 0, stream>>>(rte, mask, RT);
  k_rt_reduce<<<gr(N), 256, 0, stream>>>(bemb, mask, acc, N);
  k_head<<<1, 128, 0, stream>>>(acc, W_rt, b_rt, Wf1, bf1, Wf2, bf2, Wf3, bf3, out);
}
